// AttentionRNNLayer_77421080478285
// MI455X (gfx1250) — matrix operand order repaired and checked by reading, not compiled, not run on hardware
//
#include <hip/hip_runtime.h>

// ---------------- problem sizes ----------------
#define BATCH 64
#define TSEQ  256
#define FIN   64
#define UNITS 512
#define G4    (4*UNITS)      // 2048 gate columns
#define KTOT  (FIN+UNITS)    // 576 fused K (x | h)
#define KC    (KTOT/32)      // 18 k-chunks of 32
#define NT    (G4/16)        // 128 column tiles
#define AROWS 584            // padded A row stride (ushorts) to break LDS bank conflicts
#define TT_STRIDE ((size_t)32*KC*512*2)   // bytes between gate tiles in wsw = 589824

typedef __attribute__((ext_vector_type(16))) _Float16 v16h;
typedef __attribute__((ext_vector_type(8)))  _Float16 v8h;
typedef __attribute__((ext_vector_type(8)))  float    v8f;

static const size_t WSW_ELEMS = (size_t)NT * KC * 32 * 16;   // 1,179,648 f16 fragments
static const size_t WSW_BYTES = WSW_ELEMS * 2;               // 2,359,296 B

// -------------------------------------------------------------------------
// Kernel 1: pre-swizzle W = [kernel; rec_kernel] (576x2048 f32) into f16
// fragments in per-lane v16h operand order:
//   wsw[((nt*KC + kc)*32 + lane)*16 + j] = W[kc*32 + (l>>4)*16 + j][nt*16 + (l&15)]
// Each lane's fragment is 32 contiguous bytes -> coalesced global_load_b128 x2,
// and per-wave the (tt,kc) offsets collapse to wbase + tt*TT_STRIDE + kc*1024.
// -------------------------------------------------------------------------
__global__ void swizzle_weights(const float* __restrict__ kern,
                                const float* __restrict__ rec,
                                unsigned short* __restrict__ wsw) {
    size_t idx = (size_t)blockIdx.x * 256 + threadIdx.x;
    if (idx >= WSW_ELEMS) return;
    int j  = idx & 15;
    int l  = (idx >> 4) & 31;
    int kc = (int)((idx >> 9) % KC);
    int nt = (int)(idx / (KC * 512));
    int k  = kc * 32 + ((l >> 4) * 16) + j;
    int n  = nt * 16 + (l & 15);
    float v = (k < FIN) ? kern[(size_t)k * G4 + n]
                        : rec[(size_t)(k - FIN) * G4 + n];
    _Float16 h = (_Float16)v;
    wsw[idx] = __builtin_bit_cast(unsigned short, h);
}

// -------------------------------------------------------------------------
// Kernel 2: attention collapse coefficients.
// coef[t] = (t==0) ? 1 : sum_{j<t} att_W[t][j]; shift[t] = (t==0)?0:att_b[t]
// sumw = sum(out_W)  (folds Dense(1) over the shift term)
// -------------------------------------------------------------------------
__global__ void attn_coef(const float* __restrict__ attW,
                          const float* __restrict__ attb,
                          const float* __restrict__ outW,
                          float* __restrict__ coef,
                          float* __restrict__ shift,
                          float* __restrict__ sumw) {
    int t = threadIdx.x;                    // one block of 256
    float s = 0.f;
    for (int j = 0; j < t; ++j) s += attW[(size_t)t * TSEQ + j];
    coef[t]  = (t == 0) ? 1.0f : s;
    shift[t] = (t == 0) ? 0.0f : attb[t];
    if (t == 0) {
        float sw = 0.f;
        for (int u = 0; u < UNITS; ++u) sw += outW[u];
        *sumw = sw;
    }
}

// ---- fast activations: CDNA5 has V_TANH_F32; avoid IEEE-divide chains ----
__device__ __forceinline__ float tanh_f(float x) {
#if __has_builtin(__builtin_amdgcn_tanhf)
    return __builtin_amdgcn_tanhf(x);
#elif __has_builtin(__builtin_amdgcn_tanh_f32)
    return __builtin_amdgcn_tanh_f32(x);
#else
    float ax = fabsf(x);
    float e  = __expf(-2.0f * ax);
    float r  = (1.0f - e) * __builtin_amdgcn_rcpf(1.0f + e);
    return copysignf(r, x);
#endif
}
__device__ __forceinline__ float sigm_f(float x) {
#if __has_builtin(__builtin_amdgcn_tanhf) || __has_builtin(__builtin_amdgcn_tanh_f32)
    return 0.5f + 0.5f * tanh_f(0.5f * x);     // 1 trans + 2 VALU
#else
    return __builtin_amdgcn_rcpf(1.0f + __expf(-x));
#endif
}

// -------------------------------------------------------------------------
// Kernel 3: fused LSTM scan + attention-collapse + Dense(1) + sigmoid.
// 4 workgroups (one per 16-batch slice), 1024 threads = 32 waves.
// Wave w owns unit columns [16w, 16w+16): its 4 accumulator tiles are the
// i/f/c/o gates for those units, so the cell update is entirely in-register.
// A = [x_t | h] lives in LDS as f16; c lives in VGPRs (same thread owns the
// same cells every step). Output dot h.out_W via shfl_xor butterfly + LDS
// float atomics.
// -------------------------------------------------------------------------
__global__ __launch_bounds__(1024) void lstm_attn(
        const float* __restrict__ x,
        const unsigned short* __restrict__ wsw,
        const float* __restrict__ bias,
        const float* __restrict__ outW,
        const float* __restrict__ coef,
        const float* __restrict__ shift,
        const float* __restrict__ sumw_p,
        const float* __restrict__ outb_p,
        float* __restrict__ out) {
    __shared__ unsigned short Ash[16 * AROWS];   // rows=batch slice, cols 0..63 = x_t, 64..575 = h (f16)
    __shared__ float rowsum[16];

    const int tid  = threadIdx.x;
    const int lane = tid & 31;
    const int w    = tid >> 5;              // wave id 0..31
    const int ln15 = lane & 15;
    const int b0   = blockIdx.x * 16;
    const int u    = w * 16 + ln15;         // unit column owned in gate phase

    // ---- init: zero h region (cols 64..583), zero rowsum, zero c ----
    for (int i = tid; i < 16 * 520; i += 1024) {
        int m = i / 520, cidx = 64 + (i % 520);
        Ash[m * AROWS + cidx] = 0;
    }
    if (tid < 16) rowsum[tid] = 0.f;

    float cst[8];
#pragma unroll
    for (int r = 0; r < 8; ++r) cst[r] = 0.f;

    float bval[4];
#pragma unroll
    for (int tt = 0; tt < 4; ++tt) bval[tt] = bias[tt * UNITS + u];
    const float outw_u = outW[u];
    const float sumw   = *sumw_p;
    const float outb   = *outb_p;

    // ---- load x_0 (cols 0..63; disjoint from zeroed region) ----
    {
        int m = tid >> 6, f = tid & 63;
        float xv = x[((size_t)(b0 + m) * TSEQ + 0) * FIN + f];
        Ash[m * AROWS + f] = __builtin_bit_cast(unsigned short, (_Float16)xv);
    }

    // loop-invariant operand base pointers
    const char* wbase = (const char*)wsw + ((size_t)w * KC * 1024 + (size_t)lane * 32);
    const unsigned short* abase = &Ash[(ln15 * AROWS) + ((lane >> 4) * 16)];

    for (int t = 0; t < TSEQ; ++t) {
        __syncthreads();                    // x_t + h visible, rowsum reset done

        // ---- recurrent GEMM: g = [x_t|h] @ [kernel;rec] + bias ----
        v8f acc[4];
#pragma unroll
        for (int tt = 0; tt < 4; ++tt) {
            v8f a;
#pragma unroll
            for (int r = 0; r < 8; ++r) a[r] = bval[tt];
            acc[tt] = a;
        }
        {
            const char* wp = wbase;                  // walks +1024 B per k-chunk
            const unsigned short* ap = abase;        // walks +32 f16 per k-chunk
#pragma unroll 2
            for (int kc = 0; kc < KC; ++kc) {
                v8h alo = *(const v8h*)ap;           // k = base..base+7
                v8h ahi = *(const v8h*)(ap + 8);     // k = base+8..base+15
                v16h af = __builtin_shufflevector(alo, ahi,
                            0,1,2,3,4,5,6,7,8,9,10,11,12,13,14,15);
#pragma unroll
                for (int tt = 0; tt < 4; ++tt) {     // tiles at n = tt*512 + 16w
                    const v16h bf = *(const v16h*)(wp + (size_t)tt * TT_STRIDE);
                    acc[tt] = __builtin_amdgcn_wmma_f32_16x16x32_f16(
                            false, af, false, bf, (short)0, acc[tt], false, false);
                }
                wp += 1024;
                ap += 32;
            }
        }
        __syncthreads();                    // everyone done reading A

        // ---- in-register LSTM cell update for this wave's 16 units ----
        float p[8];
#pragma unroll
        for (int r = 0; r < 8; ++r) {
            float si = sigm_f(acc[0][r]);
            float sf = sigm_f(acc[1][r]);
            float tg = tanh_f(acc[2][r]);
            float so = sigm_f(acc[3][r]);
            float cn = sf * cst[r] + si * tg;
            cst[r] = cn;
            float h = so * tanh_f(cn);
            int row = (lane < 16) ? r : (r + 8);      // C/D layout rows
            Ash[row * AROWS + 64 + u] =
                __builtin_bit_cast(unsigned short, (_Float16)h);
            p[r] = outw_u * h;
        }
        // butterfly-reduce over the 16 lanes of each half (units 16w..16w+15)
#pragma unroll
        for (int off = 1; off < 16; off <<= 1) {
#pragma unroll
            for (int r = 0; r < 8; ++r) p[r] += __shfl_xor(p[r], off, 32);
        }
        if (ln15 == 0) {
            int rbase = (lane < 16) ? 0 : 8;
#pragma unroll
            for (int r = 0; r < 8; ++r) atomicAdd(&rowsum[rbase + r], p[r]);
        }
        __syncthreads();                    // h stores + dot atomics done

        // ---- attention-collapse + Dense(1) + sigmoid; prep next step ----
        if (tid < 16) {
            float d = rowsum[tid];
            rowsum[tid] = 0.f;
            float val = coef[t] * d + shift[t] * sumw + outb;
            out[(size_t)(b0 + tid) * TSEQ + t] = sigm_f(val);
        }
        if (t + 1 < TSEQ) {
            int m = tid >> 6, f = tid & 63;
            float xv = x[((size_t)(b0 + m) * TSEQ + (t + 1)) * FIN + f];
            Ash[m * AROWS + f] = __builtin_bit_cast(unsigned short, (_Float16)xv);
        }
    }
}

// -------------------------------------------------------------------------
extern "C" void kernel_launch(void* const* d_in, const int* in_sizes, int n_in,
                              void* d_out, int out_size, void* d_ws, size_t ws_size,
                              hipStream_t stream) {
    const float* x    = (const float*)d_in[0];
    const float* kern = (const float*)d_in[1];
    const float* rec  = (const float*)d_in[2];
    const float* bias = (const float*)d_in[3];
    const float* attW = (const float*)d_in[4];
    const float* attb = (const float*)d_in[5];
    const float* outW = (const float*)d_in[6];
    const float* outb = (const float*)d_in[7];
    float* out = (float*)d_out;

    unsigned short* wsw = (unsigned short*)d_ws;
    float* coef  = (float*)((char*)d_ws + WSW_BYTES);
    float* shift = coef + TSEQ;
    float* sumw  = shift + TSEQ;

    swizzle_weights<<<(int)((WSW_ELEMS + 255) / 256), 256, 0, stream>>>(kern, rec, wsw);
    attn_coef<<<1, TSEQ, 0, stream>>>(attW, attb, outW, coef, shift, sumw);
    lstm_attn<<<BATCH / 16, 1024, 0, stream>>>(x, wsw, bias, outW,
                                               coef, shift, sumw, outb, out);
}
